// BinaryBlock_5531917877703
// MI455X (gfx1250) — 8-bit matrix operand order repaired and checked by reading, compiled, not run on hardware
//
#include <hip/hip_runtime.h>
#include <hip/hip_bf16.h>
#include <stdint.h>

#define B_N    16
#define C_IN   64
#define C_OUT  64
#define L_N    65536
#define TILE_L 256
#define STAGE  (TILE_L + 32)

typedef __attribute__((ext_vector_type(8))) int v8i;

union Frag8 { uint4 u[2]; v8i v; };

__global__ __launch_bounds__(256) void binarize_x(const float* __restrict__ x,
                                                  const float* __restrict__ alpha,
                                                  signed char* __restrict__ xq) {
    __shared__ __align__(16) signed char t_s[64 * 64];
    const int b    = blockIdx.x >> 10;
    const int tile = blockIdx.x & 1023;
    const int l0   = tile * 64;
    const int tid  = threadIdx.x;

    #pragma unroll
    for (int it = 0; it < 4; ++it) {
        int e  = it * 1024 + tid * 4;
        int ci = e >> 6;
        int j  = e & 63;
        const float4 xv = *(const float4*)(x + ((size_t)b * C_IN + ci) * L_N + l0 + j);
        const float a = alpha[ci];
        t_s[(j + 0) * 64 + ci] = (xv.x >= a) ? (signed char)1 : (signed char)-1;
        t_s[(j + 1) * 64 + ci] = (xv.y >= a) ? (signed char)1 : (signed char)-1;
        t_s[(j + 2) * 64 + ci] = (xv.z >= a) ? (signed char)1 : (signed char)-1;
        t_s[(j + 3) * 64 + ci] = (xv.w >= a) ? (signed char)1 : (signed char)-1;
    }
    __syncthreads();
    const int off = tid * 16;
    const int row = off >> 6;
    const int c0  = off & 63;
    *(uint4*)(xq + ((size_t)b * L_N + l0 + row) * 64 + c0) = *(const uint4*)(t_s + off);
}

__global__ __launch_bounds__(256) void pack_w(const float* __restrict__ w,
                                              unsigned* __restrict__ wqp) {
    const int tid = threadIdx.x;
    #pragma unroll
    for (int i = 0; i < 12; ++i) {
        int f  = i * 256 + tid;
        int t  = f / 768;
        int r1 = f - t * 768;
        int k  = r1 / 256;
        int r2 = r1 - k * 256;
        int lane = r2 >> 3;
        int v    = r2 & 7;
        int m = lane & 15, h = lane >> 4;
        int co = t * 16 + m;
        int kb = ((v >> 2) << 5) + (h << 4) + ((v & 3) << 2);
        unsigned pk = 0;
        #pragma unroll
        for (int j = 0; j < 4; ++j) {
            int ci = kb + j;
            float wv = w[(co * C_IN + ci) * 3 + k];
            int s8 = (wv > 0.f) ? 1 : ((wv < 0.f) ? -1 : 0);
            pk |= ((unsigned)(unsigned char)(signed char)s8) << (8 * j);
        }
        wqp[f] = pk;
    }
}

__global__ __launch_bounds__(128) void bconv_wmma(const signed char* __restrict__ xq,
                                                  const unsigned*    __restrict__ wqp,
                                                  const float* __restrict__ wscale,
                                                  const float* __restrict__ bias,
                                                  const float* __restrict__ beta,
                                                  const float* __restrict__ gamma,
                                                  const float* __restrict__ zeta,
                                                  float* __restrict__ out) {
    __shared__ __align__(16) signed char xs[STAGE * 64];

    const int b     = blockIdx.x >> 8;
    const int chunk = blockIdx.x & 255;
    const int l0    = chunk * TILE_L;
    const int tid   = threadIdx.x;

    #pragma unroll
    for (int i = 0; i < 9; ++i) {
        const int off = (i * 128 + tid) * 16;
        const int row = off >> 6;
        int p  = l0 - 16 + row;
        int pc = min(max(p, 0), L_N - 1);
        unsigned long long ga =
            (unsigned long long)(uintptr_t)(xq + ((size_t)b * L_N + pc) * 64 + (off & 63));
        unsigned ldsoff = (unsigned)off;
        asm volatile("global_load_async_to_lds_b128 %0, %1, off"
                     :: "v"(ldsoff), "v"(ga) : "memory");
    }
    asm volatile("s_wait_asynccnt 0" ::: "memory");
    #pragma unroll
    for (int i = 0; i < 9; ++i) {
        const int off = (i * 128 + tid) * 16;
        const int p   = l0 - 16 + (off >> 6);
        if (p < 0 || p >= L_N) {
            uint4 z; z.x = z.y = z.z = z.w = 0u;
            *(uint4*)(xs + off) = z;
        }
    }
    __syncthreads();

    const int wv   = tid >> 5;
    const int lane = tid & 31;
    const int col  = lane & 15;
    const int hf   = lane >> 4;

    v8i afr[3];
    #pragma unroll
    for (int k = 0; k < 3; ++k) {
        const uint4* ap = (const uint4*)(wqp + ((size_t)((wv * 3 + k) * 32 + lane) * 8));
        Frag8 a; a.u[0] = ap[0]; a.u[1] = ap[1];
        afr[k] = a.v;
    }

    float ws8[8], bi8[8], be8[8], ga8[8], ze8[8];
    #pragma unroll
    for (int r = 0; r < 8; ++r) {
        int co = wv * 16 + hf * 8 + r;
        ws8[r] = wscale[co]; bi8[r] = bias[co];
        be8[r] = beta[co];   ga8[r] = gamma[co]; ze8[r] = zeta[co];
    }

    float* outb = out + ((size_t)b * C_OUT + wv * 16) * L_N + l0;

    for (int s = 0; s < TILE_L / 16; ++s) {
        v8i acc = {0, 0, 0, 0, 0, 0, 0, 0};
        #pragma unroll
        for (int k = 0; k < 3; ++k) {
            const int row = 15 + s * 16 + col + k;
            const uint4* bp = (const uint4*)(xs + row * 64 + hf * 16);
            Frag8 bu; bu.u[0] = bp[0]; bu.u[1] = bp[2];
            acc = __builtin_amdgcn_wmma_i32_16x16x64_iu8(
                true, afr[k], true, bu.v, acc, false, false);
        }
        #pragma unroll
        for (int r = 0; r < 8; ++r) {
            float y  = (float)acc[r] * ws8[r] + bi8[r];
            float xv = y - ga8[r];
            float o  = (y > ga8[r]) ? (xv + ze8[r]) : fmaf(be8[r], xv, ze8[r]);
            outb[(size_t)(hf * 8 + r) * L_N + s * 16 + col] = o;
        }
    }
}

extern "C" void kernel_launch(void* const* d_in, const int* in_sizes, int n_in,
                              void* d_out, int out_size, void* d_ws, size_t ws_size,
                              hipStream_t stream) {
    const float* x      = (const float*)d_in[0];
    const float* alpha  = (const float*)d_in[1];
    const float* weight = (const float*)d_in[2];
    const float* wscale = (const float*)d_in[3];
    const float* bias   = (const float*)d_in[4];
    const float* beta   = (const float*)d_in[5];
    const float* gamma  = (const float*)d_in[6];
    const float* zeta   = (const float*)d_in[7];

    signed char* xq = (signed char*)d_ws;
    unsigned* wqp = (unsigned*)((char*)d_ws + (size_t)B_N * L_N * C_IN);

    binarize_x<<<B_N * (L_N / 64), 256, 0, stream>>>(x, alpha, xq);
    pack_w<<<1, 256, 0, stream>>>(weight, wqp);
    bconv_wmma<<<B_N * (L_N / TILE_L), 128, 0, stream>>>(
        xq, wqp, wscale, bias, beta, gamma, zeta, (float*)d_out);
}
